// TAdapter_55327768707129
// MI455X (gfx1250) — matrix operand order repaired and checked by reading, not compiled, not run on hardware
//
#include <hip/hip_runtime.h>
#include <math.h>

// ---------------- problem sizes (fixed by the reference) ----------------
#define TB_B   8
#define TB_S   1024
#define TB_H   1024
#define TB_F   3072
#define TB_L   3
#define TB_NH  16
#define TB_DH  64
#define TB_CELLS 256

// ---------------- GEMM tiling ----------------
#define BM 128
#define BN 128
#define BK 32
#define LDA_S 40   // LDS row stride (bf16 elems), padded, keeps 16B alignment
#define LDB_S 40   // LDS B tile stored [n][k]

typedef __attribute__((ext_vector_type(8)))  float  v8f;
typedef __attribute__((ext_vector_type(8)))  __bf16 v8bf;
typedef __attribute__((ext_vector_type(16))) __bf16 v16bf;

// single packed f32x2 -> bf16x2 convert (mnemonic confirmed in gfx1250 disasm)
__device__ __forceinline__ unsigned int pack2bf(float lo, float hi) {
  unsigned int r;
  asm("v_cvt_pk_bf16_f32 %0, %1, %2" : "=v"(r) : "v"(lo), "v"(hi));
  return r;
}
__device__ __forceinline__ float gelu_f(float x) {
  return 0.5f * x * (1.0f + erff(x * 0.7071067811865476f));
}

// ============================================================================
// General bf16-WMMA GEMM:  C[b] (+)= A[b] (MxK) * B[b] (KxN) [+ bias] [gelu]
//   flags: 1=first, 2=last(apply bias/act), 4=gelu ; transB: B is (NxK)
//   batched over blockIdx.z. Double-buffered LDS, register-staged pipeline.
// ============================================================================
__global__ __launch_bounds__(256) void tb_gemm_wmma(
    const float* __restrict__ A, long lda, long strideA,
    const float* __restrict__ Bw, long ldb, long strideB,
    float* __restrict__ C, long ldc, long strideC,
    int M, int N, int K,
    const float* __restrict__ bias,
    int flags, int transB)
{
  __shared__ __align__(16) unsigned short ldsA[2][BM * LDA_S];
  __shared__ __align__(16) unsigned short ldsB[2][BN * LDB_S];

  const int tid   = threadIdx.x;
  const int lane  = tid & 31;
  const int wave  = tid >> 5;
  const int wm    = wave >> 2;     // 0..1 -> 64 rows
  const int wn    = wave & 3;      // 0..3 -> 32 cols
  const int khalf = lane >> 4;
  const int l16   = lane & 15;

  const int bm0 = blockIdx.y * BM;
  const int bn0 = blockIdx.x * BN;

  A  += (long)blockIdx.z * strideA;
  Bw += (long)blockIdx.z * strideB;
  C  += (long)blockIdx.z * strideC;

  const bool mFull = (bm0 + BM) <= M;
  const bool nFull = (bn0 + BN) <= N;

  v8f acc[4][2];
#pragma unroll
  for (int i = 0; i < 4; i++)
#pragma unroll
    for (int j = 0; j < 2; j++)
      acc[i][j] = (v8f){0.f,0.f,0.f,0.f,0.f,0.f,0.f,0.f};

  // ---- staging coordinates ----
  const int ar  = tid >> 1;            // A: row 0..127
  const int ac0 = (tid & 1) * 16;      // A: k half
  const int kr2 = (tid >> 4) * 2;      // B !trans: k-pair 0,2,..,30
  const int nb  = (tid & 15) * 8;      // B !trans: 8 contiguous n
  const int btn  = tid >> 1;           // B trans: n 0..127
  const int btc0 = (tid & 1) * 16;     // B trans: k half

  float fa[16], fb[16];

  auto loadA = [&](int k0) {
    const int  grow = bm0 + ar;
    if (mFull && (k0 + BK) <= K) {
      const float4* ap = reinterpret_cast<const float4*>(A + (long)grow * lda + k0 + ac0);
      const float4 r0 = ap[0], r1 = ap[1], r2 = ap[2], r3 = ap[3];
      fa[0]=r0.x; fa[1]=r0.y; fa[2]=r0.z; fa[3]=r0.w;
      fa[4]=r1.x; fa[5]=r1.y; fa[6]=r1.z; fa[7]=r1.w;
      fa[8]=r2.x; fa[9]=r2.y; fa[10]=r2.z; fa[11]=r2.w;
      fa[12]=r3.x; fa[13]=r3.y; fa[14]=r3.z; fa[15]=r3.w;
    } else {
      const bool rok = grow < M;
      const float* ap = A + (long)grow * lda + k0 + ac0;
#pragma unroll
      for (int i = 0; i < 16; i++) {
        const int kk = k0 + ac0 + i;
        fa[i] = (rok && kk < K) ? ap[i] : 0.0f;
      }
    }
  };

  auto loadB = [&](int k0) {
    if (!transB) {
      if (nFull && (k0 + BK) <= K) {
        const float4* p0 = reinterpret_cast<const float4*>(Bw + (long)(k0 + kr2) * ldb + bn0 + nb);
        const float4* p1 = reinterpret_cast<const float4*>(Bw + (long)(k0 + kr2 + 1) * ldb + bn0 + nb);
        const float4 a0 = p0[0], a1 = p0[1], b0 = p1[0], b1 = p1[1];
        fb[0]=a0.x; fb[1]=a0.y; fb[2]=a0.z; fb[3]=a0.w;
        fb[4]=a1.x; fb[5]=a1.y; fb[6]=a1.z; fb[7]=a1.w;
        fb[8]=b0.x; fb[9]=b0.y; fb[10]=b0.z; fb[11]=b0.w;
        fb[12]=b1.x; fb[13]=b1.y; fb[14]=b1.z; fb[15]=b1.w;
      } else {
#pragma unroll
        for (int j = 0; j < 8; j++) {
          const int n = bn0 + nb + j;
          const int k1 = k0 + kr2, k2 = k0 + kr2 + 1;
          fb[j]     = (n < N && k1 < K) ? Bw[(long)k1 * ldb + n] : 0.0f;
          fb[8 + j] = (n < N && k2 < K) ? Bw[(long)k2 * ldb + n] : 0.0f;
        }
      }
    } else {
      const int n = bn0 + btn;
      if (nFull && (k0 + BK) <= K) {
        const float4* bp = reinterpret_cast<const float4*>(Bw + (long)n * ldb + k0 + btc0);
        const float4 r0 = bp[0], r1 = bp[1], r2 = bp[2], r3 = bp[3];
        fb[0]=r0.x; fb[1]=r0.y; fb[2]=r0.z; fb[3]=r0.w;
        fb[4]=r1.x; fb[5]=r1.y; fb[6]=r1.z; fb[7]=r1.w;
        fb[8]=r2.x; fb[9]=r2.y; fb[10]=r2.z; fb[11]=r2.w;
        fb[12]=r3.x; fb[13]=r3.y; fb[14]=r3.z; fb[15]=r3.w;
      } else {
        const bool nok = n < N;
        const float* bp = Bw + (long)n * ldb + k0 + btc0;
#pragma unroll
        for (int i = 0; i < 16; i++) {
          const int kk = k0 + btc0 + i;
          fb[i] = (nok && kk < K) ? bp[i] : 0.0f;
        }
      }
    }
  };

  auto storeA = [&](unsigned short* buf) {
    uint4 v0, v1;
    v0.x = pack2bf(fa[0],  fa[1]);  v0.y = pack2bf(fa[2],  fa[3]);
    v0.z = pack2bf(fa[4],  fa[5]);  v0.w = pack2bf(fa[6],  fa[7]);
    v1.x = pack2bf(fa[8],  fa[9]);  v1.y = pack2bf(fa[10], fa[11]);
    v1.z = pack2bf(fa[12], fa[13]); v1.w = pack2bf(fa[14], fa[15]);
    *reinterpret_cast<uint4*>(&buf[ar * LDA_S + ac0])     = v0;
    *reinterpret_cast<uint4*>(&buf[ar * LDA_S + ac0 + 8]) = v1;
  };

  auto storeB = [&](unsigned short* buf) {
    if (!transB) {
#pragma unroll
      for (int j = 0; j < 8; j++) {
        const unsigned int w = pack2bf(fb[j], fb[8 + j]);
        *reinterpret_cast<unsigned int*>(&buf[(nb + j) * LDB_S + kr2]) = w;
      }
    } else {
      uint4 v0, v1;
      v0.x = pack2bf(fb[0],  fb[1]);  v0.y = pack2bf(fb[2],  fb[3]);
      v0.z = pack2bf(fb[4],  fb[5]);  v0.w = pack2bf(fb[6],  fb[7]);
      v1.x = pack2bf(fb[8],  fb[9]);  v1.y = pack2bf(fb[10], fb[11]);
      v1.z = pack2bf(fb[12], fb[13]); v1.w = pack2bf(fb[14], fb[15]);
      *reinterpret_cast<uint4*>(&buf[btn * LDB_S + btc0])     = v0;
      *reinterpret_cast<uint4*>(&buf[btn * LDB_S + btc0 + 8]) = v1;
    }
  };

  // ---- software pipeline ----
  loadA(0); loadB(0);
  int cur = 0;
  for (int k0 = 0; k0 < K; k0 += BK) {
    storeA(&ldsA[cur][0]);
    storeB(&ldsB[cur][0]);
    __syncthreads();
    if (k0 + BK < K) { loadA(k0 + BK); loadB(k0 + BK); }

    const unsigned short* bA = &ldsA[cur][0];
    const unsigned short* bB = &ldsB[cur][0];
    v16bf afr[4], bfr[2];
#pragma unroll
    for (int mi = 0; mi < 4; mi++) {
      const int row = wm * 64 + mi * 16 + l16;
      const v8bf lo = *reinterpret_cast<const v8bf*>(&bA[row * LDA_S + khalf * 8]);
      const v8bf hi = *reinterpret_cast<const v8bf*>(&bA[row * LDA_S + 16 + khalf * 8]);
      afr[mi] = __builtin_shufflevector(lo, hi, 0,1,2,3,4,5,6,7,8,9,10,11,12,13,14,15);
    }
#pragma unroll
    for (int ni = 0; ni < 2; ni++) {
      const int col = wn * 32 + ni * 16 + l16;
      const v8bf lo = *reinterpret_cast<const v8bf*>(&bB[col * LDB_S + khalf * 8]);
      const v8bf hi = *reinterpret_cast<const v8bf*>(&bB[col * LDB_S + 16 + khalf * 8]);
      bfr[ni] = __builtin_shufflevector(lo, hi, 0,1,2,3,4,5,6,7,8,9,10,11,12,13,14,15);
    }
#pragma unroll
    for (int mi = 0; mi < 4; mi++)
#pragma unroll
      for (int ni = 0; ni < 2; ni++)
        acc[mi][ni] = __builtin_amdgcn_wmma_f32_16x16x32_bf16(
            false, afr[mi], false, bfr[ni], (short)0, acc[mi][ni], false, false);
    cur ^= 1;
  }

  // ---- epilogue ----
  const bool first = (flags & 1) != 0;
  const bool last  = (flags & 2) != 0;
  const bool act   = (flags & 4) != 0;
  if (mFull && nFull) {
#pragma unroll
    for (int mi = 0; mi < 4; mi++) {
#pragma unroll
      for (int ni = 0; ni < 2; ni++) {
        const int col = bn0 + wn * 32 + ni * 16 + l16;
        const float bv = (last && bias) ? bias[col] : 0.0f;
#pragma unroll
        for (int r = 0; r < 8; r++) {
          const int row = bm0 + wm * 64 + mi * 16 + khalf * 8 + r;
          float v = acc[mi][ni][r];
          float* cp = C + (long)row * ldc + col;
          if (!first) v += *cp;
          if (last) { v += bv; if (act) v = gelu_f(v); }
          *cp = v;
        }
      }
    }
  } else {
#pragma unroll
    for (int mi = 0; mi < 4; mi++) {
#pragma unroll
      for (int ni = 0; ni < 2; ni++) {
        const int col = bn0 + wn * 32 + ni * 16 + l16;
        if (col >= N) continue;
#pragma unroll
        for (int r = 0; r < 8; r++) {
          const int row = bm0 + wm * 64 + mi * 16 + khalf * 8 + r;
          if (row >= M) continue;
          float v = acc[mi][ni][r];
          float* cp = C + (long)row * ldc + col;
          if (!first) v += *cp;
          if (last) { if (bias) v += bias[col]; if (act) v = gelu_f(v); }
          *cp = v;
        }
      }
    }
  }
}

// ============================================================================
// LayerNorm: out = LN(X + R) * g + b   (one block per row; H = 1024; float4)
// ============================================================================
__global__ __launch_bounds__(256) void tb_ln(
    const float* __restrict__ X, const float* __restrict__ Rres,
    const float* __restrict__ g, const float* __restrict__ b,
    float* __restrict__ out, int H)
{
  const long row = blockIdx.x;
  const int  t = threadIdx.x;
  const float4 xv = *reinterpret_cast<const float4*>(X + row * H + t * 4);
  const float4 rv = *reinterpret_cast<const float4*>(Rres + row * H + t * 4);
  float4 v;
  v.x = xv.x + rv.x; v.y = xv.y + rv.y; v.z = xv.z + rv.z; v.w = xv.w + rv.w;
  float s1 = v.x + v.y + v.z + v.w;
  float s2 = v.x*v.x + v.y*v.y + v.z*v.z + v.w*v.w;

  __shared__ float red1[256], red2[256];
  red1[t] = s1; red2[t] = s2; __syncthreads();
  for (int off = 128; off > 0; off >>= 1) {
    if (t < off) { red1[t] += red1[t + off]; red2[t] += red2[t + off]; }
    __syncthreads();
  }
  const float mean = red1[0] / (float)H;
  const float var  = red2[0] / (float)H - mean * mean;
  const float rstd = rsqrtf(var + 1e-12f);

  const float4 gv = *reinterpret_cast<const float4*>(g + t * 4);
  const float4 bv = *reinterpret_cast<const float4*>(b + t * 4);
  float4 o;
  o.x = (v.x - mean) * rstd * gv.x + bv.x;
  o.y = (v.y - mean) * rstd * gv.y + bv.y;
  o.z = (v.z - mean) * rstd * gv.z + bv.z;
  o.w = (v.w - mean) * rstd * gv.w + bv.w;
  *reinterpret_cast<float4*>(out + row * H + t * 4) = o;
}

// ============================================================================
// Softmax with sparse-mask additive bias (S = 1024); grid = (S, NH); float4
// ============================================================================
__global__ __launch_bounds__(256) void tb_softmax(
    float* __restrict__ sc, const float* __restrict__ mask, int S, float scale)
{
  const int q = blockIdx.x;
  const int h = blockIdx.y;
  float* row = sc + ((long)h * S + q) * (long)S;
  const float* mrow = mask + (long)q * S;
  const int t = threadIdx.x;

  const float4 rv = *reinterpret_cast<const float4*>(row + t * 4);
  const float4 mv = *reinterpret_cast<const float4*>(mrow + t * 4);
  float4 v;
  v.x = rv.x * scale + (1.0f - mv.x) * (-10000.0f);
  v.y = rv.y * scale + (1.0f - mv.y) * (-10000.0f);
  v.z = rv.z * scale + (1.0f - mv.z) * (-10000.0f);
  v.w = rv.w * scale + (1.0f - mv.w) * (-10000.0f);
  float mx = fmaxf(fmaxf(v.x, v.y), fmaxf(v.z, v.w));

  __shared__ float red[256];
  red[t] = mx; __syncthreads();
  for (int off = 128; off > 0; off >>= 1) {
    if (t < off) red[t] = fmaxf(red[t], red[t + off]);
    __syncthreads();
  }
  mx = red[0]; __syncthreads();

  v.x = expf(v.x - mx); v.y = expf(v.y - mx);
  v.z = expf(v.z - mx); v.w = expf(v.w - mx);
  red[t] = v.x + v.y + v.z + v.w; __syncthreads();
  for (int off = 128; off > 0; off >>= 1) {
    if (t < off) red[t] += red[t + off];
    __syncthreads();
  }
  const float inv = 1.0f / red[0];
  float4 o; o.x = v.x * inv; o.y = v.y * inv; o.z = v.z * inv; o.w = v.w * inv;
  *reinterpret_cast<float4*>(row + t * 4) = o;
}

// ============================================================================
// segment_sum scatter / gather (+table mask), zero — all float4
// ============================================================================
__global__ void tb_zero(float4* __restrict__ p, long n4) {
  const long i = (long)blockIdx.x * 256 + threadIdx.x;
  if (i < n4) p[i] = (float4){0.f, 0.f, 0.f, 0.f};
}

__global__ void tb_scatter(const float* __restrict__ seq, const int* __restrict__ tt,
                           int comp, float* __restrict__ cell, int Bb, int S, int H)
{
  const int H4 = H >> 2;
  const long i = (long)blockIdx.x * 256 + threadIdx.x;
  const long total = (long)Bb * S * H4;
  if (i >= total) return;
  const int h4 = (int)(i % H4) * 4;
  const long rowi = i / H4;
  const int s = (int)(rowi % S);
  const int b = (int)(rowi / S);
  const int id = tt[((long)b * S + s) * 3 + comp];
  const float4 v = *reinterpret_cast<const float4*>(seq + rowi * H + h4);
  float* cp = cell + ((long)b * TB_CELLS + id) * H + h4;
  atomicAdd(cp + 0, v.x);
  atomicAdd(cp + 1, v.y);
  atomicAdd(cp + 2, v.z);
  atomicAdd(cp + 3, v.w);
}

__global__ void tb_gather(const float* __restrict__ cell, const int* __restrict__ tt,
                          int comp, float* __restrict__ out, int Bb, int S, int H)
{
  const int H4 = H >> 2;
  const long i = (long)blockIdx.x * 256 + threadIdx.x;
  const long total = (long)Bb * S * H4;
  if (i >= total) return;
  const int h4 = (int)(i % H4) * 4;
  const long rowi = i / H4;
  const int s = (int)(rowi % S);
  const int b = (int)(rowi / S);
  const int id = tt[((long)b * S + s) * 3 + comp];
  const float m = (tt[((long)b * S + s) * 3 + 1] != 0) ? 1.0f : 0.0f;
  const float4 v = *reinterpret_cast<const float4*>(cell + ((long)b * TB_CELLS + id) * H + h4);
  float4 o; o.x = v.x * m; o.y = v.y * m; o.z = v.z * m; o.w = v.w * m;
  *reinterpret_cast<float4*>(out + rowi * H + h4) = o;
}

// ============================================================================
// host orchestration
// ============================================================================
extern "C" void kernel_launch(void* const* d_in, const int* in_sizes, int n_in,
                              void* d_out, int out_size, void* d_ws, size_t ws_size,
                              hipStream_t stream)
{
  (void)in_sizes; (void)n_in; (void)out_size; (void)ws_size;
  const int Bb = TB_B, S = TB_S, H = TB_H, F = TB_F, NH = TB_NH, DH = TB_DH;
  const long R  = (long)Bb * S;        // 8192 rows
  const long RH = R * H;               // 8,388,608 f32

  const float* emb    = (const float*)d_in[0];
  const float* allout = (const float*)d_in[1];
  const int*   tt     = (const int*)d_in[2];
  const float* spmask = (const float*)d_in[3];
  // d_in[4] attention_mask: unused by the reference
  const float* row_dp_w  = (const float*)d_in[5];
  const float* row_dp_b  = (const float*)d_in[6];
  const float* row_lin_w = (const float*)d_in[7];
  const float* row_lin_b = (const float*)d_in[8];
  const float* col_dp_w  = (const float*)d_in[9];
  const float* col_dp_b  = (const float*)d_in[10];
  const float* col_lin_w = (const float*)d_in[11];
  const float* col_lin_b = (const float*)d_in[12];
  const float* ad_dp_w   = (const float*)d_in[13];
  const float* ad_dp_b   = (const float*)d_in[14];
  const float* ad_ln_g   = (const float*)d_in[15];
  const float* ad_ln_b   = (const float*)d_in[16];
  const float* q_w = (const float*)d_in[17]; const float* q_b = (const float*)d_in[18];
  const float* k_w = (const float*)d_in[19]; const float* k_b = (const float*)d_in[20];
  const float* v_w = (const float*)d_in[21]; const float* v_b = (const float*)d_in[22];
  const float* o_w = (const float*)d_in[23]; const float* o_b = (const float*)d_in[24];
  const float* attn_ln_g = (const float*)d_in[25];
  const float* attn_ln_b = (const float*)d_in[26];
  const float* ff1_w = (const float*)d_in[27]; const float* ff1_b = (const float*)d_in[28];
  const float* ff2_w = (const float*)d_in[29]; const float* ff2_b = (const float*)d_in[30];
  const float* ff_ln_g = (const float*)d_in[31];
  const float* ff_ln_b = (const float*)d_in[32];

  // ---- workspace layout (overlayed; peak = 10*RH f32 = 336 MB) ----
  float* ws     = (float*)d_ws;
  float* former = ws;                 // RH
  float* T1     = ws + 1 * RH;        // RH
  float* Xb     = ws + 2 * RH;        // RH
  float* Yb     = ws + 3 * RH;        // RH
  float* REG2   = ws + 4 * RH;        // 3*RH : {row,col,cell,cell2} then {Q,K,V}
  float* REG1   = ws + 7 * RH;        // 3*RH : {score,ctx} then FF
  float* rowb  = REG2;
  float* colb  = REG2 + RH;
  float* cellb = REG2 + 2 * RH;
  float* cell2 = REG2 + 2 * RH + (long)Bb * TB_CELLS * H;
  float* Qb = REG2;
  float* Kb = REG2 + RH;
  float* Vb = REG2 + 2 * RH;
  float* score = REG1;                 // NH*S*S per-b chunk
  float* ctx   = REG1 + 2 * RH;
  float* FF    = REG1;                 // R*F after attention done

  auto gemm = [&](const float* A, long lda, long sA,
                  const float* Bm, long ldb, long sB,
                  float* Cm, long ldc, long sC,
                  int M, int N, int K, const float* bias,
                  int flags, int transB, int batch) {
    dim3 grid((N + BN - 1) / BN, (M + BM - 1) / BM, batch);
    tb_gemm_wmma<<<grid, dim3(256), 0, stream>>>(A, lda, sA, Bm, ldb, sB,
                                                 Cm, ldc, sC, M, N, K,
                                                 bias, flags, transB);
  };

  const long tot4   = RH / 4;
  const long cell4  = ((long)Bb * TB_CELLS * H) / 4;
  const int  GRID_T4 = (int)((tot4 + 255) / 256);
  const int  GRID_C4 = (int)((cell4 + 255) / 256);

  hipMemcpyAsync(former, emb, RH * sizeof(float), hipMemcpyDeviceToDevice, stream);

  for (int lx = 0; lx < TB_L; lx++) {
    const float* hidden = allout + (long)lx * RH;

    // ---------- table-wise (row then col) ----------
    for (int tw = 0; tw < 2; tw++) {
      const float* dpw  = (tw == 0 ? row_dp_w  : col_dp_w)  + (long)lx * 2 * H * H;
      const float* dpb  = (tw == 0 ? row_dp_b  : col_dp_b)  + (long)lx * H;
      const float* linw = (tw == 0 ? row_lin_w : col_lin_w) + (long)lx * H * H;
      const float* linb = (tw == 0 ? row_lin_b : col_lin_b) + (long)lx * H;
      const int comp = (tw == 0) ? 2 : 1;
      float* outb = (tw == 0) ? rowb : colb;

      gemm(former, H, 0, dpw,             H, 0, T1, H, 0, (int)R, H, H, nullptr, 1,     0, 1);
      gemm(hidden, H, 0, dpw + (long)H*H, H, 0, T1, H, 0, (int)R, H, H, dpb,     2 | 4, 0, 1);
      tb_zero<<<GRID_C4, 256, 0, stream>>>((float4*)cellb, cell4);
      tb_scatter<<<GRID_T4, 256, 0, stream>>>(T1, tt, comp, cellb, Bb, S, H);
      gemm(cellb, H, 0, linw, H, 0, cell2, H, 0, Bb * TB_CELLS, H, H, linb, 1 | 2 | 4, 0, 1);
      tb_gather<<<GRID_T4, 256, 0, stream>>>(cell2, tt, comp, outb, Bb, S, H);
    }

    // ---------- adapter ----------
    {
      const float* adw = ad_dp_w + (long)lx * 4 * H * H;
      const float* adb = ad_dp_b + (long)lx * H;
      gemm(former, H, 0, adw,              H, 0, T1, H, 0, (int)R, H, H, nullptr, 1,     0, 1);
      gemm(hidden, H, 0, adw + 1L * H * H, H, 0, T1, H, 0, (int)R, H, H, nullptr, 0,     0, 1);
      gemm(rowb,   H, 0, adw + 2L * H * H, H, 0, T1, H, 0, (int)R, H, H, nullptr, 0,     0, 1);
      gemm(colb,   H, 0, adw + 3L * H * H, H, 0, T1, H, 0, (int)R, H, H, adb,     2 | 4, 0, 1);
      tb_ln<<<(int)R, 256, 0, stream>>>(T1, former,
                                        ad_ln_g + (long)lx * H, ad_ln_b + (long)lx * H,
                                        Xb, H);
    }

    // ---------- attention ----------
    gemm(Xb, H, 0, q_w + (long)lx * H * H, H, 0, Qb, H, 0, (int)R, H, H, q_b + (long)lx * H, 1 | 2, 0, 1);
    gemm(Xb, H, 0, k_w + (long)lx * H * H, H, 0, Kb, H, 0, (int)R, H, H, k_b + (long)lx * H, 1 | 2, 0, 1);
    gemm(Xb, H, 0, v_w + (long)lx * H * H, H, 0, Vb, H, 0, (int)R, H, H, v_b + (long)lx * H, 1 | 2, 0, 1);

    for (int b = 0; b < Bb; b++) {
      gemm(Qb + (long)b * S * H, H, DH,
           Kb + (long)b * S * H, H, DH,
           score, S, (long)S * S,
           S, S, DH, nullptr, 1 | 2, 1 /*transB*/, NH);
      tb_softmax<<<dim3(S, NH), 256, 0, stream>>>(score, spmask + (long)b * S * S, S, 0.125f);
      gemm(score, S, (long)S * S,
           Vb + (long)b * S * H, H, DH,
           ctx + (long)b * S * H, H, DH,
           S, DH, S, nullptr, 1 | 2, 0, NH);
    }

    gemm(ctx, H, 0, o_w + (long)lx * H * H, H, 0, T1, H, 0, (int)R, H, H, o_b + (long)lx * H, 1 | 2, 0, 1);
    tb_ln<<<(int)R, 256, 0, stream>>>(T1, Xb,
                                      attn_ln_g + (long)lx * H, attn_ln_b + (long)lx * H,
                                      Yb, H);

    // ---------- FFN ----------
    gemm(Yb, H, 0, ff1_w + (long)lx * H * F, F, 0, FF, F, 0, (int)R, F, H,
         ff1_b + (long)lx * F, 1 | 2 | 4, 0, 1);
    gemm(FF, F, 0, ff2_w + (long)lx * F * H, H, 0, T1, H, 0, (int)R, H, F,
         ff2_b + (long)lx * H, 1 | 2, 0, 1);
    tb_ln<<<(int)R, 256, 0, stream>>>(T1, Yb,
                                      ff_ln_g + (long)lx * H, ff_ln_b + (long)lx * H,
                                      former, H);
  }

  hipMemcpyAsync(d_out, former, RH * sizeof(float), hipMemcpyDeviceToDevice, stream);
}
